// GCN_51780125721117
// MI455X (gfx1250) — matrix operand order repaired and checked by reading, not compiled, not run on hardware
//
#include <hip/hip_runtime.h>
#include <cstdint>
#include <cstddef>

#define NN 40000
#define EE 640000
#define BB 64
#define INDIM 64
#define EDIM 16

typedef __attribute__((ext_vector_type(16))) _Float16 v16h;
typedef __attribute__((ext_vector_type(8)))  float    v8f;

// ---------------------------------------------------------------------------
// WMMA tiled GEMM:  OUT[M,Nc] = A[M,K] @ W[K,Nc] + bias   (row-major, f32 io)
// K compile-time. Per block (8 waves): W is staged once into LDS, converted to
// f16 and PRE-SWIZZLED into the exact WMMA B-fragment lane layout, so each
// lane's 16 B halves are contiguous (two ds_load_b128 per K-chunk). Zero
// padding for K=16 is baked into the LDS image (branch-free compute loop).
// Each wave owns a 16-row stripe, keeps all A fragments in registers, and
// sweeps every 16-column tile.  v_wmma_f32_16x16x32_f16, wave32.
// ---------------------------------------------------------------------------
template<int K>
__global__ void k_gemm_bias(const float* __restrict__ A,
                            const float* __restrict__ W,
                            const float* __restrict__ bias,
                            float* __restrict__ OUT,
                            int M, int Nc)
{
    constexpr int KC = (K + 31) / 32;          // K=32 WMMA steps
    extern __shared__ _Float16 ldsW[];         // [tilesN][KC][32 lanes][16 halves]

    const int lane = threadIdx.x & 31;
    const int wave = blockIdx.x * (blockDim.x >> 5) + (threadIdx.x >> 5);
    const int tilesM = M >> 4;
    const int tilesN = Nc >> 4;

    // ---- cooperative stage of W into B-fragment layout ----
    // idx -> (tn, kc, lane, j):  frag element j of lane, chunk kc, col-tile tn
    const int total = tilesN * KC * 512;       // halves
    for (int idx = threadIdx.x; idx < total; idx += blockDim.x) {
        const int j  = idx & 15;
        const int ln = (idx >> 4) & 31;
        const int kt = idx >> 9;               // tn*KC + kc
        const int kc = kt % KC;
        const int tn = kt / KC;
        const int k  = kc * 32 + ((ln >> 4) << 3) + (j & 7) + ((j >> 3) << 4);   // kofB in {0,8}, +16 for j>=8
        const int col = (tn << 4) + (ln & 15);
        ldsW[idx] = (k < K) ? (_Float16)W[(size_t)k * Nc + col] : (_Float16)0.f;
    }
    __syncthreads();

    if (wave < tilesM) {
        const int tm = wave << 4;
        // A 16x32 f16 layout: lanes 0-15 -> row=lane, K 0..7 & 16..23;
        //                     lanes 16-31 -> same row, K 8..15 & 24..31
        const int rowA = tm + (lane & 15);
        const int kofA = (lane >> 4) << 3;     // 0 or 8

        const float* arow = A + (size_t)rowA * K;
        v16h afrag[KC];
#pragma unroll
        for (int kc = 0; kc < KC; ++kc) {
            const int k0 = kc * 32;
            const float4 p0 = *(const float4*)(arow + k0 + kofA);
            const float4 p1 = *(const float4*)(arow + k0 + kofA + 4);
            afrag[kc][0] = (_Float16)p0.x; afrag[kc][1] = (_Float16)p0.y;
            afrag[kc][2] = (_Float16)p0.z; afrag[kc][3] = (_Float16)p0.w;
            afrag[kc][4] = (_Float16)p1.x; afrag[kc][5] = (_Float16)p1.y;
            afrag[kc][6] = (_Float16)p1.z; afrag[kc][7] = (_Float16)p1.w;
            if (k0 + 16 < K) {                 // compile-time after unroll
                const float4 p2 = *(const float4*)(arow + k0 + 16 + kofA);
                const float4 p3 = *(const float4*)(arow + k0 + 20 + kofA);
                afrag[kc][8]  = (_Float16)p2.x; afrag[kc][9]  = (_Float16)p2.y;
                afrag[kc][10] = (_Float16)p2.z; afrag[kc][11] = (_Float16)p2.w;
                afrag[kc][12] = (_Float16)p3.x; afrag[kc][13] = (_Float16)p3.y;
                afrag[kc][14] = (_Float16)p3.z; afrag[kc][15] = (_Float16)p3.w;
            } else {
#pragma unroll
                for (int i = 8; i < 16; ++i) afrag[kc][i] = (_Float16)0.f;
            }
        }

        const int colOff = lane & 15;
        const int rbase  = tm + ((lane >> 4) << 3);
        for (int tn = 0; tn < tilesN; ++tn) {
            const int col = (tn << 4) + colOff;
            v8f acc = {};
#pragma unroll
            for (int kc = 0; kc < KC; ++kc) {
                // lane's B fragment: 16 contiguous halves -> 2x ds_load_b128
                const v16h b = *(const v16h*)&ldsW[(((tn * KC + kc) << 5) + lane) << 4];
                // (neg_a, A, neg_b, B, c_mod, C, reuse_a, reuse_b)
                acc = __builtin_amdgcn_wmma_f32_16x16x32_f16(false, afrag[kc], false, b,
                                                             (short)0, acc, false, false);
            }
            const float bi = bias[col];
#pragma unroll
            for (int r = 0; r < 8; ++r)
                OUT[(size_t)(rbase + r) * Nc + col] = acc[r] + bi;
        }
    }
}

// ---------------------------------------------------------------------------
// Edge path: msg = relu(xsrc[src] + elin) + 1e-7 ; segment max via int-bits
// (valid: all msgs > 0). msg overwrites elin in place.
// ---------------------------------------------------------------------------
__global__ void k_msg_max(const float* __restrict__ xsrc,
                          float* __restrict__ msg,          // in: elin, out: msg
                          const int* __restrict__ src,
                          const int* __restrict__ dst,
                          float* __restrict__ mx,
                          int C)
{
    const int i = blockIdx.x * blockDim.x + threadIdx.x;
    if (i >= EE * C) return;
    const int e = i / C, c = i - e * C;
    float v = xsrc[(size_t)src[e] * C + c] + msg[i];
    v = (v > 0.f ? v : 0.f) + 1e-7f;
    msg[i] = v;
    atomicMax((int*)&mx[(size_t)dst[e] * C + c], __float_as_int(v));
}

__global__ void k_exp_den(const float* __restrict__ msg,
                          const int* __restrict__ dst,
                          const float* __restrict__ mx,
                          float* __restrict__ den, int C)
{
    const int i = blockIdx.x * blockDim.x + threadIdx.x;
    if (i >= EE * C) return;
    const int e = i / C, c = i - e * C;
    const size_t d = (size_t)dst[e] * C + c;
    atomicAdd(&den[d], __expf(msg[i] - mx[d]));
}

__global__ void k_agg(const float* __restrict__ msg,
                      const int* __restrict__ dst,
                      const float* __restrict__ mx,
                      const float* __restrict__ den,
                      float* __restrict__ agg, int C)
{
    const int i = blockIdx.x * blockDim.x + threadIdx.x;
    if (i >= EE * C) return;
    const int e = i / C, c = i - e * C;
    const size_t d = (size_t)dst[e] * C + c;
    const float m  = msg[i];
    const float al = __expf(m - mx[d]) / (den[d] + 1e-16f);
    atomicAdd(&agg[d], m * al);
}

// out[n,c] += add[n,c]   (xdst += agg)
__global__ void k_add_inplace(float* __restrict__ out,
                              const float* __restrict__ add, int total)
{
    const int i = blockIdx.x * blockDim.x + threadIdx.x;
    if (i < total) out[i] += add[i];
}

// ---------------------------------------------------------------------------
// BatchNorm (training mode, biased variance). One block per channel: LDS tree.
// mv[0..C) = mean, mv[C..2C) = var
// ---------------------------------------------------------------------------
__global__ void k_bn_stats(const float* __restrict__ x,
                           float* __restrict__ mv, int M, int C)
{
    const int c = blockIdx.x;
    float s = 0.f, s2 = 0.f;
    for (int r = threadIdx.x; r < M; r += blockDim.x) {
        const float v = x[(size_t)r * C + c];
        s += v; s2 += v * v;
    }
    __shared__ float sh[256], sh2[256];
    sh[threadIdx.x] = s; sh2[threadIdx.x] = s2;
    __syncthreads();
    for (int off = blockDim.x >> 1; off > 0; off >>= 1) {
        if (threadIdx.x < (unsigned)off) {
            sh[threadIdx.x]  += sh[threadIdx.x + off];
            sh2[threadIdx.x] += sh2[threadIdx.x + off];
        }
        __syncthreads();
    }
    if (threadIdx.x == 0) {
        const float mu = sh[0] / (float)M;
        mv[c]     = mu;
        mv[C + c] = sh2[0] / (float)M - mu * mu;
    }
}

// mode: 0=none, 1=relu, 2=leaky(0.01). In place.
__global__ void k_bn_apply(float* __restrict__ x,
                           const float* __restrict__ mv,
                           const float* __restrict__ g,
                           const float* __restrict__ be,
                           int M, int C, int mode)
{
    const int i = blockIdx.x * blockDim.x + threadIdx.x;
    if (i >= M * C) return;
    const int c = i % C;
    float v = (x[i] - mv[c]) * rsqrtf(mv[C + c] + 1e-5f) * g[c] + be[c];
    if (mode == 1)      v = v > 0.f ? v : 0.f;
    else if (mode == 2) v = v > 0.f ? v : 0.01f * v;
    x[i] = v;
}

// ---------------------------------------------------------------------------
// global_mean_pool
// ---------------------------------------------------------------------------
__global__ void k_pool_sum(const float* __restrict__ h,
                           const int* __restrict__ batch,
                           float* __restrict__ out, int C)
{
    const int i = blockIdx.x * blockDim.x + threadIdx.x;
    if (i >= NN * C) return;
    const int n = i / C, c = i - n * C;
    atomicAdd(&out[(size_t)batch[n] * C + c], h[i]);
}

__global__ void k_pool_cnt(const int* __restrict__ batch, float* __restrict__ cnt)
{
    const int i = blockIdx.x * blockDim.x + threadIdx.x;
    if (i < NN) atomicAdd(&cnt[batch[i]], 1.f);
}

__global__ void k_pool_div(float* __restrict__ out,
                           const float* __restrict__ cnt, int C)
{
    const int i = blockIdx.x * blockDim.x + threadIdx.x;
    if (i >= BB * C) return;
    const float c = cnt[i / C];
    out[i] /= (c > 1.f ? c : 1.f);
}

// ---------------------------------------------------------------------------
// Host orchestration
// ---------------------------------------------------------------------------
static inline int ceil_div(int a, int b) { return (a + b - 1) / b; }

static void launch_gemm(int K, const float* A, const float* W, const float* b,
                        float* O, int M, int Nc, hipStream_t s)
{
    const int blocks = ceil_div(M >> 4, 8);    // 8 waves (256 threads) per block
    const int KC = (K + 31) / 32;
    const size_t sh = (size_t)(Nc >> 4) * KC * 512 * sizeof(_Float16); // <= 64KB
    switch (K) {
        case 16:  k_gemm_bias<16> <<<blocks, 256, sh, s>>>(A, W, b, O, M, Nc); break;
        case 32:  k_gemm_bias<32> <<<blocks, 256, sh, s>>>(A, W, b, O, M, Nc); break;
        case 64:  k_gemm_bias<64> <<<blocks, 256, sh, s>>>(A, W, b, O, M, Nc); break;
        case 128: k_gemm_bias<128><<<blocks, 256, sh, s>>>(A, W, b, O, M, Nc); break;
        case 256: k_gemm_bias<256><<<blocks, 256, sh, s>>>(A, W, b, O, M, Nc); break;
        default: break;
    }
}

extern "C" void kernel_launch(void* const* d_in, const int* in_sizes, int n_in,
                              void* d_out, int out_size, void* d_ws, size_t ws_size,
                              hipStream_t stream)
{
    (void)in_sizes; (void)ws_size; (void)out_size;

    const float* x     = (const float*)d_in[0];
    const int*   ei    = (const int*)  d_in[1];
    const float* eattr = (const float*)d_in[2];
    const int*   batch = (const int*)  d_in[3];
    const int*   src   = ei;
    const int*   dst   = ei + EE;

    // Flattened param leaves, setup_inputs() dict order:
    // conv{k}: lin_src.W,b lin_dst.W,b lin_edge.W,b mlp1.W,b mlp_bn.g,be mlp2.W,b
    // then bn{k}.g, bn{k}.be
    const float* P[64];
    for (int i = 0; i < 64; ++i) P[i] = (i < n_in) ? (const float*)d_in[i] : nullptr;
    const int convBase[3] = {4, 18, 32};
    const int bnBase[3]   = {16, 30, 44};
    const int cins[3]  = {INDIM, 32, 64};
    const int couts[3] = {32, 64, 128};

    // ---- workspace carve (256B aligned) ----
    char* w = (char*)d_ws;
    auto carve = [&](size_t bytes) -> float* {
        float* p = (float*)w;
        w += (bytes + 255) & ~(size_t)255;
        return p;
    };
    const size_t NC  = (size_t)NN * 128 * sizeof(float);
    const size_t N2C = (size_t)NN * 256 * sizeof(float);
    const size_t EC  = (size_t)EE * 128 * sizeof(float);
    float* xsrc  = carve(NC);
    float* xdst  = carve(NC);
    float* elin  = carve(EC);                 // -> becomes msg in place
    float* mx    = carve(NC);
    float* den   = carve(NC);
    float* agg   = carve(NC);
    float* h2    = carve(N2C);
    float* hA    = carve(NC);
    float* hB    = carve(NC);
    float* mv    = carve(2 * 256 * sizeof(float));
    float* cnt   = carve(BB * sizeof(float));

    const int TB = 256;
    const float* h = x;                        // layer input
    float* hbuf[2] = {hA, hB};

    for (int L = 0; L < 3; ++L) {
        const int cb = convBase[L], bb = bnBase[L];
        const float *Ws = P[cb+0], *bs = P[cb+1];
        const float *Wd = P[cb+2], *bd = P[cb+3];
        const float *We = P[cb+4], *bpe = P[cb+5];
        const float *W1 = P[cb+6], *b1 = P[cb+7];
        const float *g1 = P[cb+8], *be1 = P[cb+9];
        const float *W2 = P[cb+10], *b2 = P[cb+11];
        const float *gO = P[bb+0],  *beO = P[bb+1];
        const int cin = cins[L], cout = couts[L];
        float* hout = hbuf[L & 1];

        // dense projections (WMMA)
        launch_gemm(cin, h, Ws, bs, xsrc, NN, cout, stream);
        launch_gemm(cin, h, Wd, bd, xdst, NN, cout, stream);
        launch_gemm(EDIM, eattr, We, bpe, elin, EE, cout, stream);

        // softmax aggregation over edges (L2-resident accumulators)
        const size_t nc = (size_t)NN * cout * sizeof(float);
        hipMemsetAsync(mx,  0, nc, stream);
        hipMemsetAsync(den, 0, nc, stream);
        hipMemsetAsync(agg, 0, nc, stream);
        const int ecTot = EE * cout;
        k_msg_max<<<ceil_div(ecTot, TB), TB, 0, stream>>>(xsrc, elin, src, dst, mx, cout);
        k_exp_den<<<ceil_div(ecTot, TB), TB, 0, stream>>>(elin, dst, mx, den, cout);
        k_agg    <<<ceil_div(ecTot, TB), TB, 0, stream>>>(elin, dst, mx, den, agg, cout);

        // out = agg + lin_dst(x)   (into xdst)
        k_add_inplace<<<ceil_div(NN * cout, TB), TB, 0, stream>>>(xdst, agg, NN * cout);

        // GENConv MLP: Linear(C,2C) -> BN -> ReLU -> Linear(2C,C)
        launch_gemm(cout, xdst, W1, b1, h2, NN, 2 * cout, stream);
        k_bn_stats<<<2 * cout, TB, 0, stream>>>(h2, mv, NN, 2 * cout);
        k_bn_apply<<<ceil_div(NN * 2 * cout, TB), TB, 0, stream>>>(h2, mv, g1, be1, NN, 2 * cout, 1);
        launch_gemm(2 * cout, h2, W2, b2, hout, NN, cout, stream);

        // outer BN + LeakyReLU(0.01)
        k_bn_stats<<<cout, TB, 0, stream>>>(hout, mv, NN, cout);
        k_bn_apply<<<ceil_div(NN * cout, TB), TB, 0, stream>>>(hout, mv, gO, beO, NN, cout, 2);

        h = hout;
    }

    // global mean pool over batch -> d_out [BB, 128]
    float* out = (float*)d_out;
    const int C3 = 128;
    hipMemsetAsync(out, 0, (size_t)BB * C3 * sizeof(float), stream);
    hipMemsetAsync(cnt, 0, BB * sizeof(float), stream);
    k_pool_sum<<<ceil_div(NN * C3, TB), TB, 0, stream>>>(h, batch, out, C3);
    k_pool_cnt<<<ceil_div(NN, TB), TB, 0, stream>>>(batch, cnt);
    k_pool_div<<<ceil_div(BB * C3, TB), TB, 0, stream>>>(out, cnt, C3);
}
